// Kernel_Layer_2284922601517
// MI455X (gfx1250) — compile-verified, matrix operand order checked by reading, not run on hardware
//
#include <hip/hip_runtime.h>
#include <stdint.h>

typedef __attribute__((ext_vector_type(16))) __bf16 v16bf;
typedef __attribute__((ext_vector_type(8)))  float  v8f;

union FragAB {
  v16bf v;
  uint4 q[2];
};

__device__ __forceinline__ uint16_t f32_to_bf16_rne(float f) {
  uint32_t u = __float_as_uint(f);
  u += 0x7FFFu + ((u >> 16) & 1u);   // round-to-nearest-even
  return (uint16_t)(u >> 16);
}

__device__ __forceinline__ float bf16_bits_to_f32(uint16_t h) {
  return __uint_as_float(((uint32_t)h) << 16);
}

// ---------------------------------------------------------------------------
// Kernel 1: split f32 rows of x / prototypes into bf16 (hi, lo) pairs and
// compute per-row squared norms.  One 256-thread block per row (D == 256).
// ---------------------------------------------------------------------------
__global__ void rbf_split_norms(const float* __restrict__ x,
                                const float* __restrict__ prot,
                                uint16_t* __restrict__ xhi, uint16_t* __restrict__ xlo,
                                uint16_t* __restrict__ phi, uint16_t* __restrict__ plo,
                                float* __restrict__ xsq, float* __restrict__ psq) {
  const int row = blockIdx.x;            // 0 .. 8191
  const int t   = threadIdx.x;           // 0 .. 255 (== D)
  const bool isX = row < 4096;
  const int r = isX ? row : row - 4096;

  const float* src = (isX ? x : prot) + (size_t)r * 256;
  uint16_t* hi = (isX ? xhi : phi) + (size_t)r * 256;
  uint16_t* lo = (isX ? xlo : plo) + (size_t)r * 256;

  float f = src[t];
  uint16_t hb = f32_to_bf16_rne(f);
  float hf = bf16_bits_to_f32(hb);
  uint16_t lb = f32_to_bf16_rne(f - hf);
  hi[t] = hb;
  lo[t] = lb;

  __shared__ float red[256];
  red[t] = f * f;
  __syncthreads();
#pragma unroll
  for (int s = 128; s > 0; s >>= 1) {
    if (t < s) red[t] += red[t + s];
    __syncthreads();
  }
  if (t == 0) (isX ? xsq : psq)[r] = red[0];
}

// ---------------------------------------------------------------------------
// Fragment loader: for v_wmma_f32_16x16x32_bf16, lane L holds row (L&15),
// K-halves selected by (L>>4): VGPRs 0-3 = K [khalf .. khalf+7],
// VGPRs 4-7 = K [16+khalf .. 16+khalf+7].  Two b128 loads, fully coalesced.
// Same pattern serves both A (x rows) and B (prototype rows, since B is
// consumed K-major == prototypes row-major).
// ---------------------------------------------------------------------------
__device__ __forceinline__ v16bf load_frag(const uint16_t* __restrict__ base,
                                           int row, int kLane) {
  const uint16_t* p = base + (size_t)row * 256 + kLane;
  FragAB f;
  f.q[0] = *(const uint4*)(p);
  f.q[1] = *(const uint4*)(p + 16);
  return f.v;
}

// ---------------------------------------------------------------------------
// Kernel 2: bf16 split-GEMM + fused RBF epilogue.
// Block = 256 threads = 8 waves (2 M-waves x 4 N-waves).
// Wave tile = 64x64 (16 accumulators), workgroup tile = 128x256.
// cross ~= xhi*phi + xhi*plo + xlo*phi  (fp32-class accuracy).
// ---------------------------------------------------------------------------
__global__ __launch_bounds__(256)
void rbf_gemm_wmma(const uint16_t* __restrict__ xhi, const uint16_t* __restrict__ xlo,
                   const uint16_t* __restrict__ phi, const uint16_t* __restrict__ plo,
                   const float* __restrict__ xsq, const float* __restrict__ psq,
                   const int* __restrict__ sigmaPtr,
                   float* __restrict__ out) {
  const int lane  = threadIdx.x & 31;
  const int wave  = threadIdx.x >> 5;         // 0..7
  const int wm    = wave >> 2;                // 0..1
  const int wn    = wave & 3;                 // 0..3
  const int rowBase = blockIdx.y * 128 + wm * 64;
  const int colBase = blockIdx.x * 256 + wn * 64;
  const int mrow  = lane & 15;                // fragment row / C column N
  const int khalf = (lane >> 4) << 3;         // 0 or 8

  v8f acc[4][4];
#pragma unroll
  for (int mt = 0; mt < 4; ++mt)
#pragma unroll
    for (int nt = 0; nt < 4; ++nt)
      acc[mt][nt] = (v8f){0.f, 0.f, 0.f, 0.f, 0.f, 0.f, 0.f, 0.f};

  for (int k0 = 0; k0 < 256; k0 += 32) {
    // Prefetch next K-slice for this wave's A rows (global_prefetch_b8).
    if (k0 + 32 < 256) {
      __builtin_prefetch(xhi + (size_t)(rowBase + mrow) * 256 + k0 + 32 + khalf, 0, 0);
      __builtin_prefetch(phi + (size_t)(colBase + mrow) * 256 + k0 + 32 + khalf, 0, 0);
    }

    v16bf aHi[4], aLo[4];
#pragma unroll
    for (int mt = 0; mt < 4; ++mt) {
      int rA = rowBase + mt * 16 + mrow;
      aHi[mt] = load_frag(xhi, rA, k0 + khalf);
      aLo[mt] = load_frag(xlo, rA, k0 + khalf);
    }

#pragma unroll
    for (int nt = 0; nt < 4; ++nt) {
      int rB = colBase + nt * 16 + mrow;
      v16bf bHi = load_frag(phi, rB, k0 + khalf);
      v16bf bLo = load_frag(plo, rB, k0 + khalf);
#pragma unroll
      for (int mt = 0; mt < 4; ++mt) {
        acc[mt][nt] = __builtin_amdgcn_wmma_f32_16x16x32_bf16(
            false, aHi[mt], false, bHi, (short)0, acc[mt][nt], false, false);
        acc[mt][nt] = __builtin_amdgcn_wmma_f32_16x16x32_bf16(
            false, aHi[mt], false, bLo, (short)0, acc[mt][nt], false, false);
        acc[mt][nt] = __builtin_amdgcn_wmma_f32_16x16x32_bf16(
            false, aLo[mt], false, bHi, (short)0, acc[mt][nt], false, false);
      }
    }
  }

  // Fused RBF epilogue: d2 = |x|^2 + |p|^2 - 2*cross ; out = exp(-d2/(2s^2)).
  const float s = (float)sigmaPtr[0];
  const float inv2s2 = 1.0f / (2.0f * s * s);

#pragma unroll
  for (int mt = 0; mt < 4; ++mt) {
    // C/D layout: VGPR r of lane L -> M = r + 8*(L>>4), N = L & 15.
    const int rb = rowBase + mt * 16 + khalf;
    float xs[8];
#pragma unroll
    for (int r = 0; r < 8; ++r) xs[r] = xsq[rb + r];
#pragma unroll
    for (int nt = 0; nt < 4; ++nt) {
      const int col = colBase + nt * 16 + mrow;
      const float ps = psq[col];
#pragma unroll
      for (int r = 0; r < 8; ++r) {
        float d2 = xs[r] + ps - 2.0f * acc[mt][nt][r];
        out[(size_t)(rb + r) * 4096 + col] = __expf(-d2 * inv2s2);
      }
    }
  }
}

// ---------------------------------------------------------------------------
// Host-side launcher.
// Workspace layout (8.02 MB used):
//   [0,2MB)      xhi   bf16[4096][256]
//   [2MB,4MB)    xlo
//   [4MB,6MB)    phi
//   [6MB,8MB)    plo
//   [8MB, +16KB) xsq   f32[4096]
//   [+16KB,+32KB) psq  f32[4096]
// ---------------------------------------------------------------------------
extern "C" void kernel_launch(void* const* d_in, const int* in_sizes, int n_in,
                              void* d_out, int out_size, void* d_ws, size_t ws_size,
                              hipStream_t stream) {
  const float* x    = (const float*)d_in[0];
  const float* prot = (const float*)d_in[1];
  const int*   sig  = (const int*)d_in[2];
  float*       out  = (float*)d_out;

  uint8_t* ws = (uint8_t*)d_ws;
  const size_t MB = 1024u * 1024u;
  uint16_t* xhi = (uint16_t*)(ws + 0 * MB);
  uint16_t* xlo = (uint16_t*)(ws + 2 * MB);
  uint16_t* phi = (uint16_t*)(ws + 4 * MB);
  uint16_t* plo = (uint16_t*)(ws + 6 * MB);
  float*    xsq = (float*)(ws + 8 * MB);
  float*    psq = (float*)(ws + 8 * MB + 16 * 1024);

  rbf_split_norms<<<8192, 256, 0, stream>>>(x, prot, xhi, xlo, phi, plo, xsq, psq);

  dim3 grid(4096 / 256, 4096 / 128);   // (N tiles, M tiles) = (16, 32)
  rbf_gemm_wmma<<<grid, 256, 0, stream>>>(xhi, xlo, phi, plo, xsq, psq, sig, out);
}
